// VisionReformerLMStem_21414706938670
// MI455X (gfx1250) — compile-verified, matrix operand order checked by reading, not run on hardware
//
#include <hip/hip_runtime.h>
#include <math.h>

// ---------------- types ----------------
typedef __attribute__((ext_vector_type(16))) __bf16 v16bf;
typedef __attribute__((ext_vector_type(8)))  float  v8f;

#define IMG   560
#define PIXN  (8*560*560)
#define TOK   784            // tokens per image
#define MROWS 6272           // 8*784
#define DIM   256
#define FF    1024
#define DH    32
#define NB    56             // buckets / chunks
#define BK    14             // bucket size

// ---------------- small helpers ----------------
__device__ __forceinline__ float gelu_exact(float x) {
    return 0.5f * x * (1.0f + erff(x * 0.70710678118654752f));
}
__device__ __forceinline__ v8f wmma_bf16(v16bf a, v16bf b, v8f c) {
    return __builtin_amdgcn_wmma_f32_16x16x32_bf16(false, a, false, b, (short)0, c, false, false);
}

// ---------------- BN fold ----------------
__global__ void bn_prep_kernel(const float* g, const float* b, const float* m,
                               const float* v, float* scale, float* shift, int C) {
    int i = blockIdx.x * blockDim.x + threadIdx.x;
    if (i >= C) return;
    float s = g[i] * rsqrtf(v[i] + 1e-5f);
    scale[i] = s;
    shift[i] = b[i] - m[i] * s;
}

// ---------------- conv1 (3->32) direct + BN + GELU, NHWC bf16 out ----------------
__global__ void conv1_kernel(const float* __restrict__ x, const float* __restrict__ w,
                             const float* __restrict__ scale, const float* __restrict__ shift,
                             __bf16* __restrict__ out) {
    int idx = blockIdx.x * blockDim.x + threadIdx.x;
    if (idx >= PIXN) return;
    int px = idx % IMG, py = (idx / IMG) % IMG, n = idx / (IMG * IMG);
    float in[27];
    #pragma unroll
    for (int c = 0; c < 3; ++c)
        #pragma unroll
        for (int kh = 0; kh < 3; ++kh)
            #pragma unroll
            for (int kw = 0; kw < 3; ++kw) {
                int y = py + kh - 1, xx = px + kw - 1;
                in[c * 9 + kh * 3 + kw] =
                    (y >= 0 && y < IMG && xx >= 0 && xx < IMG)
                        ? x[((size_t)(n * 3 + c) * IMG + y) * IMG + xx] : 0.0f;
            }
    for (int o = 0; o < 32; ++o) {
        float acc = 0.f;
        #pragma unroll
        for (int j = 0; j < 27; ++j) acc += in[j] * w[o * 27 + j];
        float v = acc * scale[o] + shift[o];
        out[(size_t)idx * 32 + o] = (__bf16)gelu_exact(v);
    }
}

// ---------------- weight repacks (to [N][K] bf16, K 16-contiguous per lane) ------
__global__ void repack_conv2w_kernel(const float* src, __bf16* dst) { // (64,32,3,3)
    int i = blockIdx.x * blockDim.x + threadIdx.x;
    if (i >= 64 * 288) return;
    int o = i / 288, k = i % 288, tap = k / 32, ci = k % 32;
    int kh = tap / 3, kw = tap % 3;
    dst[i] = (__bf16)src[((o * 32 + ci) * 3 + kh) * 3 + kw];
}
__global__ void repack_patchw_kernel(const float* src, __bf16* dst) { // (256,64,20,20)
    int i = blockIdx.x * blockDim.x + threadIdx.x;
    if (i >= 256 * 25600) return;
    int o = i / 25600, k = i % 25600, pp = k / 64, ci = k % 64;
    int py = pp / 20, px = pp % 20;
    dst[i] = (__bf16)src[((o * 64 + ci) * 20 + py) * 20 + px];
}
__global__ void transpose_bf16_kernel(const float* src, __bf16* dst, int K, int N) {
    // src [K][N] f32 -> dst [N][K] bf16
    int i = blockIdx.x * blockDim.x + threadIdx.x;
    if (i >= K * N) return;
    int n = i / K, k = i % K;
    dst[i] = (__bf16)src[(size_t)k * N + n];
}

// ---------------- conv2 (32->64) implicit-GEMM WMMA + BN + GELU -----------------
// One wave = 16 pixels x 64 channels (4 accumulators share one A fragment).
// in: NHWC bf16 (8,560,560,32); w: [64][288] bf16 k=(tap*32+ci); out: NHWC bf16 x64
__global__ void conv2_wmma_kernel(const __bf16* __restrict__ in, const __bf16* __restrict__ w,
                                  const float* __restrict__ scale, const float* __restrict__ shift,
                                  __bf16* __restrict__ out) {
    int wv   = (blockIdx.x * blockDim.x + threadIdx.x) >> 5;   // m-tile id, 8*560*35 total
    int lane = threadIdx.x & 31;
    int xg = wv % 35, y = (wv / 35) % IMG, n = wv / (35 * IMG);
    int px = xg * 16 + (lane & 15);
    int kh16 = (lane >> 4) * 16;                 // K-half select (contiguous k = 16h + s, same for A and B)
    const __bf16* wrow = w + (size_t)(lane & 15) * 288 + kh16;
    v8f acc[4] = {};
    bool interior = (y > 0) & (y < IMG - 1) & (xg > 0) & (xg < 34);  // wave-uniform
    if (interior) {
        #pragma unroll
        for (int tap = 0; tap < 9; ++tap) {
            int kh = tap / 3, kw = tap % 3;
            int yy = y + kh - 1, xx = px + kw - 1;
            v16bf a = *(const v16bf*)(in + (((size_t)n * IMG + yy) * IMG + xx) * 32 + kh16);
            #pragma unroll
            for (int nt = 0; nt < 4; ++nt) {
                v16bf b = *(const v16bf*)(wrow + (size_t)nt * 16 * 288 + tap * 32);
                acc[nt] = wmma_bf16(a, b, acc[nt]);
            }
        }
    } else {
        #pragma unroll
        for (int tap = 0; tap < 9; ++tap) {
            int kh = tap / 3, kw = tap % 3;
            int yy = y + kh - 1, xx = px + kw - 1;
            v16bf a = {};
            if (yy >= 0 && yy < IMG && xx >= 0 && xx < IMG)
                a = *(const v16bf*)(in + (((size_t)n * IMG + yy) * IMG + xx) * 32 + kh16);
            #pragma unroll
            for (int nt = 0; nt < 4; ++nt) {
                v16bf b = *(const v16bf*)(wrow + (size_t)nt * 16 * 288 + tap * 32);
                acc[nt] = wmma_bf16(a, b, acc[nt]);
            }
        }
    }
    int rbase = (lane >> 4) * 8;
    #pragma unroll
    for (int nt = 0; nt < 4; ++nt) {
        int c = nt * 16 + (lane & 15);
        float sc = scale[c], sh = shift[c];
        #pragma unroll
        for (int r = 0; r < 8; ++r) {
            int xo = xg * 16 + r + rbase;
            float v = acc[nt][r] * sc + sh;
            out[(((size_t)n * IMG + y) * IMG + xo) * 64 + c] = (__bf16)gelu_exact(v);
        }
    }
}

// ---------------- patch embed WMMA: M=6272, N=256, K=25600 ----------------------
// One wave = 16 patches x 64 dims (4 accumulators share one A fragment).
__global__ void patch_wmma_kernel(const __bf16* __restrict__ in, const __bf16* __restrict__ wp,
                                  const float* __restrict__ pb, const float* __restrict__ pos,
                                  float* __restrict__ x1, float* __restrict__ x2) {
    int wv   = (blockIdx.x * blockDim.x + threadIdx.x) >> 5;  // 392*4 waves
    int lane = threadIdx.x & 31;
    int mt = wv >> 2, ng = wv & 3;               // m-tile, 64-col group
    int mrow = mt * 16 + (lane & 15);
    int b = mrow / TOK, t = mrow % TOK, ty = t / 28, tx = t % 28;
    int kh16 = (lane >> 4) * 16;
    const __bf16* wrow = wp + (size_t)(ng * 64 + (lane & 15)) * 25600 + kh16;
    v8f acc[4] = {};
    for (int pp = 0; pp < 400; ++pp) {
        int py = pp / 20, px = pp % 20;
        int y = ty * 20 + py, x = tx * 20 + px;
        const __bf16* base = in + (((size_t)b * IMG + y) * IMG + x) * 64;
        __builtin_prefetch((const void*)(base + 64 * 20), 0, 3);   // same pixel, next patch along x
        #pragma unroll
        for (int cg = 0; cg < 2; ++cg) {
            v16bf a = *(const v16bf*)(base + cg * 32 + kh16);
            #pragma unroll
            for (int nt = 0; nt < 4; ++nt) {
                v16bf bb = *(const v16bf*)(wrow + (size_t)nt * 16 * 25600 + pp * 64 + cg * 32);
                acc[nt] = wmma_bf16(a, bb, acc[nt]);
            }
        }
    }
    int rbase = (lane >> 4) * 8;
    #pragma unroll
    for (int nt = 0; nt < 4; ++nt) {
        int col = ng * 64 + nt * 16 + (lane & 15);
        float bv = pb[col];
        #pragma unroll
        for (int r = 0; r < 8; ++r) {
            int row = mt * 16 + r + rbase;
            float v = acc[nt][r] + bv + pos[(size_t)(row % TOK) * DIM + col];
            x1[(size_t)row * DIM + col] = v;
            x2[(size_t)row * DIM + col] = v;
        }
    }
}

// ---------------- generic WMMA GEMM: C = act(A[M,K] * Bt[N,K]^T + bias (+res)) ---
// One wave = 16 rows x 64 cols; N must be a multiple of 64.
__global__ void wmma_gemm_kernel(const __bf16* __restrict__ A, const __bf16* __restrict__ Bt,
                                 const float* __restrict__ bias, const float* __restrict__ residual,
                                 float* __restrict__ Cf, __bf16* __restrict__ Cb,
                                 int M, int N, int K, int act) {
    int wave = (blockIdx.x * blockDim.x + threadIdx.x) >> 5;
    int lane = threadIdx.x & 31;
    int ngroups = N >> 6;
    int tiles = (M >> 4) * ngroups;
    if (wave >= tiles) return;                   // wave-uniform: EXEC stays full
    int tm = wave / ngroups, tg = wave % ngroups;
    int kh16 = (lane >> 4) * 16;
    const __bf16* arow = A  + (size_t)(tm * 16 + (lane & 15)) * K + kh16;
    const __bf16* brow = Bt + (size_t)(tg * 64 + (lane & 15)) * K + kh16;
    v8f acc[4] = {};
    for (int k0 = 0; k0 < K; k0 += 32) {
        __builtin_prefetch((const void*)(arow + k0 + 64), 0, 3);
        v16bf a = *(const v16bf*)(arow + k0);
        #pragma unroll
        for (int nt = 0; nt < 4; ++nt) {
            v16bf b = *(const v16bf*)(brow + (size_t)nt * 16 * K + k0);
            acc[nt] = wmma_bf16(a, b, acc[nt]);
        }
    }
    int rbase = (lane >> 4) * 8;
    #pragma unroll
    for (int nt = 0; nt < 4; ++nt) {
        int col = tg * 64 + nt * 16 + (lane & 15);
        float bv = bias ? bias[col] : 0.0f;
        #pragma unroll
        for (int r = 0; r < 8; ++r) {
            int row = tm * 16 + r + rbase;
            size_t idx = (size_t)row * N + col;
            float v = acc[nt][r] + bv;
            if (residual) v += residual[idx];
            if (act) v = gelu_exact(v);
            if (Cf) Cf[idx] = v;
            if (Cb) Cb[idx] = (__bf16)v;
        }
    }
}

// ---------------- LayerNorm (wave per row) -> bf16 ------------------------------
__global__ void ln_bf16_kernel(const float* __restrict__ x, const float* __restrict__ g,
                               const float* __restrict__ b, __bf16* __restrict__ out, int rows) {
    int wave = (blockIdx.x * blockDim.x + threadIdx.x) >> 5;
    int lane = threadIdx.x & 31;
    if (wave >= rows) return;
    const float* r = x + (size_t)wave * DIM;
    float vals[8], s = 0.f;
    #pragma unroll
    for (int i = 0; i < 8; ++i) { vals[i] = r[lane + i * 32]; s += vals[i]; }
    #pragma unroll
    for (int off = 16; off; off >>= 1) s += __shfl_xor(s, off, 32);
    float mean = s * (1.0f / DIM), s2 = 0.f;
    #pragma unroll
    for (int i = 0; i < 8; ++i) { float d = vals[i] - mean; s2 += d * d; }
    #pragma unroll
    for (int off = 16; off; off >>= 1) s2 += __shfl_xor(s2, off, 32);
    float inv = rsqrtf(s2 * (1.0f / DIM) + 1e-5f);
    #pragma unroll
    for (int i = 0; i < 8; ++i) {
        int c = lane + i * 32;
        out[(size_t)wave * DIM + c] = (__bf16)((vals[i] - mean) * inv * g[c] + b[c]);
    }
}

// ---------------- LSH bucketing: argmax over [R, -R], rotations in LDS ----------
__global__ void bucket_kernel(const float* __restrict__ qk, const float* __restrict__ rot,
                              int* __restrict__ buckets) {
    __shared__ float srot[DH * 28];
    for (int i = threadIdx.x; i < DH * 28; i += blockDim.x) srot[i] = rot[i];
    __syncthreads();
    int idx = blockIdx.x * blockDim.x + threadIdx.x;   // 64*784
    if (idx >= 64 * TOK) return;
    int t = idx % TOK, bh = idx / TOK, b = bh / 8, h = bh % 8;
    const float* q = qk + (size_t)(b * TOK + t) * DIM + h * DH;
    float qv[DH];
    #pragma unroll
    for (int d = 0; d < DH; ++d) qv[d] = q[d];
    float best = -1e30f; int bi = 0;
    for (int i = 0; i < 28; ++i) {
        float s = 0.f;
        #pragma unroll
        for (int d = 0; d < DH; ++d) s += qv[d] * srot[d * 28 + i];
        if (s > best)  { best = s;  bi = i; }
        if (-s > best) { best = -s; bi = i + 28; }
    }
    buckets[idx] = bi;
}

// ---------------- stable counting sort per (b,h): thread-per-bucket -------------
__global__ void sort_kernel(const int* __restrict__ buckets, int* __restrict__ sticker) {
    __shared__ int cnt[NB + 1];
    int bh = blockIdx.x, j = threadIdx.x;              // 64 threads
    const int* bk = buckets + bh * TOK;
    if (j == 0) cnt[0] = 0;
    if (j < NB) {
        int c = 0;
        for (int t = 0; t < TOK; ++t) c += (bk[t] == j);
        cnt[j + 1] = c;
    }
    __syncthreads();
    if (j == 0) for (int i = 1; i <= NB; ++i) cnt[i] += cnt[i - 1];
    __syncthreads();
    if (j < NB) {
        int pos = cnt[j];
        for (int t = 0; t < TOK; ++t)
            if (bk[t] == j) sticker[bh * TOK + pos++] = t;
    }
}

// ---------------- chunked LSH attention (14 q x 28 kv, look-one-back) -----------
__global__ void chunk_attn_kernel(const float* __restrict__ qk, const float* __restrict__ v,
                                  const int* __restrict__ sticker, __bf16* __restrict__ attn_out) {
    __shared__ float q[BK][DH], kn[2 * BK][DH], vv[2 * BK][DH];
    __shared__ float dots[BK][2 * BK];
    __shared__ int tq[BK], tk[2 * BK];
    int blk = blockIdx.x, bh = blk / NB, c = blk % NB;
    int b = bh / 8, h = bh % 8;
    int lane = threadIdx.x;                            // 32 threads
    int cprev = (c + NB - 1) % NB;
    if (lane < BK) { int t = sticker[bh * TOK + c * BK + lane]; tq[lane] = t; tk[lane] = t; }
    else if (lane < 2 * BK) tk[lane] = sticker[bh * TOK + cprev * BK + (lane - BK)];
    __syncthreads();
    for (int i = 0; i < BK; ++i)
        q[i][lane] = qk[(size_t)(b * TOK + tq[i]) * DIM + h * DH + lane];
    for (int j = 0; j < 2 * BK; ++j) {
        int t = tk[j];
        kn[j][lane] = qk[(size_t)(b * TOK + t) * DIM + h * DH + lane];
        vv[j][lane] = v [(size_t)(b * TOK + t) * DIM + h * DH + lane];
    }
    __syncthreads();
    if (lane < 2 * BK) {                               // F.normalize the keys
        float s = 0.f;
        #pragma unroll
        for (int d = 0; d < DH; ++d) { float x = kn[lane][d]; s += x * x; }
        float inv = 1.0f / fmaxf(sqrtf(s), 1e-12f);
        #pragma unroll
        for (int d = 0; d < DH; ++d) kn[lane][d] *= inv;
    }
    __syncthreads();
    for (int p = lane; p < BK * 2 * BK; p += 32) {
        int i = p / (2 * BK), j = p % (2 * BK);
        float s = 0.f;
        #pragma unroll
        for (int d = 0; d < DH; ++d) s += q[i][d] * kn[j][d];
        s *= 0.17677669529663687f;                     // 32^-0.5
        if (tq[i] == tk[j]) s = -5e4f;                 // self-attn mask
        dots[i][j] = s;
    }
    __syncthreads();
    for (int i = 0; i < BK; ++i) {                     // softmax + PV, dim = lane
        float m = -1e30f;
        #pragma unroll
        for (int j = 0; j < 2 * BK; ++j) m = fmaxf(m, dots[i][j]);
        float s = 0.f, o = 0.f;
        #pragma unroll
        for (int j = 0; j < 2 * BK; ++j) {
            float e = __expf(dots[i][j] - m);
            s += e; o += e * vv[j][lane];
        }
        attn_out[(size_t)(b * TOK + tq[i]) * DIM + h * DH + lane] = (__bf16)(o / s);
    }
}

// ---------------- final: mean over tokens, LN, head -----------------------------
__global__ void final_head_kernel(const float* __restrict__ x1, const float* __restrict__ x2,
                                  const float* __restrict__ g, const float* __restrict__ bb,
                                  const float* __restrict__ hw, const float* __restrict__ hb,
                                  float* __restrict__ out) {
    __shared__ float col[DIM], tmp[DIM];
    int b = blockIdx.x, d = threadIdx.x;               // 256 threads
    float s = 0.f;
    for (int t = 0; t < TOK; ++t) {
        size_t idx = (size_t)(b * TOK + t) * DIM + d;
        s += 0.5f * (x1[idx] + x2[idx]);
    }
    s *= (1.0f / TOK);
    tmp[d] = s; __syncthreads();
    for (int off = 128; off; off >>= 1) { if (d < off) tmp[d] += tmp[d + off]; __syncthreads(); }
    float mean = tmp[0] * (1.0f / DIM);
    float dv = s - mean;
    __syncthreads();
    tmp[d] = dv * dv; __syncthreads();
    for (int off = 128; off; off >>= 1) { if (d < off) tmp[d] += tmp[d + off]; __syncthreads(); }
    float inv = rsqrtf(tmp[0] * (1.0f / DIM) + 1e-5f);
    col[d] = dv * inv * g[d] + bb[d];
    __syncthreads();
    if (d < 5) {
        float acc = hb[d];
        for (int k = 0; k < DIM; ++k) acc += col[k] * hw[k * 5 + d];
        out[b * 5 + d] = acc;
    }
}

// =================================================================================
extern "C" void kernel_launch(void* const* d_in, const int* in_sizes, int n_in,
                              void* d_out, int out_size, void* d_ws, size_t ws_size,
                              hipStream_t stream) {
    (void)in_sizes; (void)n_in; (void)out_size; (void)ws_size;
    const float* x       = (const float*)d_in[0];
    const float* conv1_w = (const float*)d_in[1];
    const float* bn1_g = (const float*)d_in[2], *bn1_b = (const float*)d_in[3];
    const float* bn1_m = (const float*)d_in[4], *bn1_v = (const float*)d_in[5];
    const float* conv2_w = (const float*)d_in[6];
    const float* bn2_g = (const float*)d_in[7], *bn2_b = (const float*)d_in[8];
    const float* bn2_m = (const float*)d_in[9], *bn2_v = (const float*)d_in[10];
    const float* patch_w = (const float*)d_in[11];
    const float* patch_b = (const float*)d_in[12];
    const float* pos     = (const float*)d_in[13];
    const float* rots    = (const float*)d_in[14];
    const float* Wqk = (const float*)d_in[15], *Wv = (const float*)d_in[16];
    const float* Wo  = (const float*)d_in[17], *bo = (const float*)d_in[18];
    const float* ln1_g = (const float*)d_in[19], *ln1_b = (const float*)d_in[20];
    const float* W1 = (const float*)d_in[21], *b1 = (const float*)d_in[22];
    const float* W2 = (const float*)d_in[23], *b2 = (const float*)d_in[24];
    const float* ln2_g = (const float*)d_in[25], *ln2_b = (const float*)d_in[26];
    const float* lnf_g = (const float*)d_in[27], *lnf_b = (const float*)d_in[28];
    const float* head_w = (const float*)d_in[29], *head_b = (const float*)d_in[30];
    float* out = (float*)d_out;

    // ---- workspace carve-out (256B aligned) ----
    char* ws = (char*)d_ws;
    size_t off = 0;
    auto carve = [&](size_t bytes) { size_t p = off; off = (off + bytes + 255) & ~(size_t)255; return p; };
    __bf16* conv1o = (__bf16*)(ws + carve((size_t)PIXN * 32 * 2));
    __bf16* conv2o = (__bf16*)(ws + carve((size_t)PIXN * 64 * 2));
    __bf16* w2r    = (__bf16*)(ws + carve((size_t)64 * 288 * 2));
    __bf16* wpr    = (__bf16*)(ws + carve((size_t)256 * 25600 * 2));
    __bf16* wlr    = (__bf16*)(ws + carve((size_t)8 * 720896 * 2));
    float*  x1     = (float*)(ws + carve((size_t)MROWS * DIM * 4));
    float*  x2     = (float*)(ws + carve((size_t)MROWS * DIM * 4));
    float*  qkf    = (float*)(ws + carve((size_t)MROWS * DIM * 4));
    float*  vf     = (float*)(ws + carve((size_t)MROWS * DIM * 4));
    __bf16* lnbf   = (__bf16*)(ws + carve((size_t)MROWS * DIM * 2));
    __bf16* ffbf   = (__bf16*)(ws + carve((size_t)MROWS * FF * 2));
    __bf16* attnbf = (__bf16*)(ws + carve((size_t)MROWS * DIM * 2));
    int*    buckets= (int*)(ws + carve((size_t)64 * TOK * 4));
    int*    sticker= (int*)(ws + carve((size_t)64 * TOK * 4));
    float*  bn1s   = (float*)(ws + carve(32 * 4));
    float*  bn1t   = (float*)(ws + carve(32 * 4));
    float*  bn2s   = (float*)(ws + carve(64 * 4));
    float*  bn2t   = (float*)(ws + carve(64 * 4));

    // ---- BN folds + weight repacks ----
    bn_prep_kernel<<<1, 64, 0, stream>>>(bn1_g, bn1_b, bn1_m, bn1_v, bn1s, bn1t, 32);
    bn_prep_kernel<<<1, 64, 0, stream>>>(bn2_g, bn2_b, bn2_m, bn2_v, bn2s, bn2t, 64);
    repack_conv2w_kernel<<<(64 * 288 + 255) / 256, 256, 0, stream>>>(conv2_w, w2r);
    repack_patchw_kernel<<<(256 * 25600 + 255) / 256, 256, 0, stream>>>(patch_w, wpr);
    for (int l = 0; l < 8; ++l) {
        __bf16* base = wlr + (size_t)l * 720896;
        transpose_bf16_kernel<<<(65536 + 255) / 256, 256, 0, stream>>>(Wqk + (size_t)l * 65536, base, 256, 256);
        transpose_bf16_kernel<<<(65536 + 255) / 256, 256, 0, stream>>>(Wv  + (size_t)l * 65536, base + 65536, 256, 256);
        transpose_bf16_kernel<<<(65536 + 255) / 256, 256, 0, stream>>>(Wo  + (size_t)l * 65536, base + 131072, 256, 256);
        transpose_bf16_kernel<<<(262144 + 255) / 256, 256, 0, stream>>>(W1 + (size_t)l * 262144, base + 196608, 256, 1024);
        transpose_bf16_kernel<<<(262144 + 255) / 256, 256, 0, stream>>>(W2 + (size_t)l * 262144, base + 458752, 1024, 256);
    }

    // ---- stem ----
    conv1_kernel<<<(PIXN + 255) / 256, 256, 0, stream>>>(x, conv1_w, bn1s, bn1t, conv1o);
    conv2_wmma_kernel<<<(8 * 560 * 35) / 8, 256, 0, stream>>>(conv1o, w2r, bn2s, bn2t, conv2o);
    patch_wmma_kernel<<<(392 * 4) / 8, 256, 0, stream>>>(conv2o, wpr, patch_b, pos, x1, x2);

    // ---- reversible Reformer layers ----
    auto gemm = [&](const __bf16* A, const __bf16* Bt, const float* bias, const float* res,
                    float* Cf, __bf16* Cb, int M, int N, int K, int act) {
        int tiles = (M / 16) * (N / 64);
        wmma_gemm_kernel<<<(tiles + 7) / 8, 256, 0, stream>>>(A, Bt, bias, res, Cf, Cb, M, N, K, act);
    };
    for (int l = 0; l < 8; ++l) {
        const __bf16* wqkT = wlr + (size_t)l * 720896;
        const __bf16* wvT  = wqkT + 65536;
        const __bf16* woT  = wqkT + 131072;
        const __bf16* w1T  = wqkT + 196608;
        const __bf16* w2T  = wqkT + 458752;
        // x1 = x1 + attn(LN(x2))
        ln_bf16_kernel<<<MROWS / 8, 256, 0, stream>>>(x2, ln1_g + l * DIM, ln1_b + l * DIM, lnbf, MROWS);
        gemm(lnbf, wqkT, nullptr, nullptr, qkf, nullptr, MROWS, DIM, DIM, 0);
        gemm(lnbf, wvT,  nullptr, nullptr, vf,  nullptr, MROWS, DIM, DIM, 0);
        bucket_kernel<<<(64 * TOK + 255) / 256, 256, 0, stream>>>(qkf, rots + (size_t)l * DH * 28, buckets);
        sort_kernel<<<64, 64, 0, stream>>>(buckets, sticker);
        chunk_attn_kernel<<<64 * NB, 32, 0, stream>>>(qkf, vf, sticker, attnbf);
        gemm(attnbf, woT, bo + l * DIM, x1, x1, nullptr, MROWS, DIM, DIM, 0);
        // x2 = x2 + ff(LN(x1))
        ln_bf16_kernel<<<MROWS / 8, 256, 0, stream>>>(x1, ln2_g + l * DIM, ln2_b + l * DIM, lnbf, MROWS);
        gemm(lnbf, w1T, b1 + l * FF, nullptr, nullptr, ffbf, MROWS, FF, DIM, 1);
        gemm(ffbf, w2T, b2 + l * DIM, x2, x2, nullptr, MROWS, DIM, FF, 0);
    }

    // ---- final head ----
    final_head_kernel<<<8, 256, 0, stream>>>(x1, x2, lnf_g, lnf_b, head_w, head_b, out);
}
